// SelfAttention2d_51934744543300
// MI455X (gfx1250) — compile-verified, matrix operand order checked by reading, not run on hardware
//
#include <hip/hip_runtime.h>
#include <math.h>

#define BATCH  16
#define CCH    512
#define SEQ    1024
#define THREEC 1536
#define GN_EPS 1e-5f
#define QSCALE 0.044194173824159216f   // 512^-0.5
#define LDST   40                      // LDS row stride in halves (32 data + 8 pad = 80B)

typedef __attribute__((ext_vector_type(16))) _Float16 v16h;
typedef __attribute__((ext_vector_type(8)))  _Float16 v8h;
typedef __attribute__((ext_vector_type(8)))  float    v8f;
typedef __attribute__((ext_vector_type(4)))  unsigned int u32x4;
typedef __attribute__((ext_vector_type(8)))  int      i32x8;
typedef __attribute__((ext_vector_type(4)))  int      i32x4;

#if defined(__AMDGCN__) && __has_builtin(__builtin_amdgcn_tensor_load_to_lds)
#define USE_TDM 1
#else
#define USE_TDM 0
#endif

// ---------------- GroupNorm stats (1 group): per-batch mean / rstd ----------------
__global__ void gn_stats_kernel(const float* __restrict__ x, float* __restrict__ stats) {
  const int b = blockIdx.x;
  const int t = threadIdx.x;
  const float4* xp = (const float4*)(x + (size_t)b * CCH * SEQ);
  const int n4 = (CCH * SEQ) / 4;
  float s = 0.f, ss = 0.f;
  for (int i = t; i < n4; i += 256) {
    float4 v = xp[i];
    s  += v.x + v.y + v.z + v.w;
    ss += v.x*v.x + v.y*v.y + v.z*v.z + v.w*v.w;
  }
  __shared__ float rs[256], rss[256];
  rs[t] = s; rss[t] = ss; __syncthreads();
  for (int o = 128; o > 0; o >>= 1) {
    if (t < o) { rs[t] += rs[t+o]; rss[t] += rss[t+o]; }
    __syncthreads();
  }
  if (t == 0) {
    const float inv = 1.f / (float)(CCH * SEQ);
    float mean = rs[0] * inv;
    float var  = rss[0] * inv - mean * mean;
    stats[2*b]   = mean;
    stats[2*b+1] = rsqrtf(var + GN_EPS);
  }
}

// ---------------- fp32 -> fp16 weight conversion ----------------
__global__ void convert_w_kernel(const float* __restrict__ qkvw, const float* __restrict__ projw,
                                 _Float16* __restrict__ wq, _Float16* __restrict__ wp) {
  int i = blockIdx.x * 256 + threadIdx.x;
  const int nq = THREEC * CCH;
  if (i < nq) wq[i] = (_Float16)qkvw[i];
  int j = i - nq;
  if (j >= 0 && j < CCH * CCH) wp[j] = (_Float16)projw[j];
}

// -------- normalize + scale/shift + transpose [C,S] -> [S,C] fp16 (LDS tiled) --------
__global__ void normalize_kernel(const float* __restrict__ x, const float* __restrict__ gw,
                                 const float* __restrict__ gb, const float* __restrict__ stats,
                                 _Float16* __restrict__ ht, int b) {
  __shared__ float tile[32][33];
  const int c0 = blockIdx.x * 32, s0 = blockIdx.y * 32;
  const int tx = threadIdx.x, ty = threadIdx.y;
  const float mean = stats[2*b], rstd = stats[2*b+1];
  const float* xb = x + (size_t)b * CCH * SEQ;
  for (int i = 0; i < 4; i++) {
    int c = c0 + ty + i*8;
    tile[ty + i*8][tx] = xb[(size_t)c * SEQ + s0 + tx];
  }
  __syncthreads();
  for (int i = 0; i < 4; i++) {
    int s = s0 + ty + i*8;
    int c = c0 + tx;
    float v = (tile[tx][ty + i*8] - mean) * rstd * gw[c] + gb[c];
    ht[(size_t)s * CCH + c] = (_Float16)v;
  }
}

// ---------------- TDM: DMA a 128x32 fp16 tile (row stride K) into padded LDS ----------------
#if USE_TDM
__device__ __forceinline__ unsigned lds_off32(const void* p) {
  // generic LDS pointer: low 32 bits are the LDS byte address
  return (unsigned)(unsigned long long)(size_t)p;
}

__device__ __forceinline__ void tdm_load_tile_128x32(const _Float16* gptr, unsigned ldsAddr, int K) {
  unsigned long long ga = (unsigned long long)(size_t)gptr;
  u32x4 g0;
  g0[0] = 1u;                                   // count=1, user descriptor, no gather
  g0[1] = ldsAddr;                              // lds_addr (bytes)
  g0[2] = (unsigned)(ga & 0xFFFFFFFFull);       // global_addr[31:0]
  g0[3] = (unsigned)((ga >> 32) & 0x1FFFFFFull) // global_addr[56:32]
        | (2u << 30);                           // type = 2 (image)
  i32x8 g1;
  g1[0] = (1 << 16)                             // data_size = 2 bytes
        | (1 << 20)                             // pad_enable
        | (3 << 22)                             // pad_interval: every 16 DWORDs (64B row)
        | (3 << 25);                            // pad_amount: 4 DWORDs (16B) -> LDST=40 halves
  g1[1] = (K & 0xFFFF) << 16;                   // tensor_dim0[15:0] = K
  g1[2] = ((K >> 16) & 0xFFFF) | (128 << 16);   // tensor_dim0[31:16] | tensor_dim1[15:0]=128
  g1[3] = (32 << 16);                           // tensor_dim1 hi | tile_dim0 = 32
  g1[4] = 128;                                  // tile_dim1 = 128, tile_dim2 = 0
  g1[5] = K;                                    // tensor_dim0_stride[31:0] = K elements
  g1[6] = 0;                                    // stride0 hi | stride1 lo
  g1[7] = 0;
  const i32x4 z4 = {0, 0, 0, 0};
  const i32x8 z8 = {0, 0, 0, 0, 0, 0, 0, 0};
  // 6-arg toolchain variant: (g0, g1, g2, g3, g4, cpol)
  __builtin_amdgcn_tensor_load_to_lds(g0, g1, z4, z4, z8, 0);
}
#endif

// ---------------- WMMA GEMM: D = A[M,K] * B[N,K]^T, fp16 in / fp32 acc ----------------
// 128x128 block tile, 8 waves, each wave 32x64 (2x4 tiles of 16x16), K-step 32.
// Double-buffered LDS fed by the Tensor Data Mover (wave 0 issues, s_wait_tensorcnt paces).
enum { MODE_QKV = 0, MODE_SCORES = 1, MODE_PV = 2, MODE_PROJ = 3 };

template<int MODE>
__global__ void __launch_bounds__(256)
gemm_kernel(const _Float16* __restrict__ A, const _Float16* __restrict__ Bm, int K,
            const float* __restrict__ bias,
            _Float16* __restrict__ o16a, _Float16* __restrict__ o16b, _Float16* __restrict__ o16c,
            float* __restrict__ ofl, const float* __restrict__ resid) {
  __shared__ _Float16 As[2][128 * LDST];
  __shared__ _Float16 Bs[2][128 * LDST];

  const int t    = threadIdx.x;
  const int lane = t & 31;
  const int wave = t >> 5;
  const int wm   = wave & 3;   // 4 row groups of 32
  const int wn   = wave >> 2;  // 2 col groups of 64
  const int blockN = blockIdx.x * 128;
  const int blockM = blockIdx.y * 128;

  v8f acc[2][4];
  const v8f vzero = {0.f,0.f,0.f,0.f,0.f,0.f,0.f,0.f};
  for (int i = 0; i < 2; i++)
    for (int j = 0; j < 4; j++)
      acc[i][j] = vzero;

  const int nIter = K >> 5;
  const _Float16* gA = A  + (size_t)blockM * K;
  const _Float16* gB = Bm + (size_t)blockN * K;

#if USE_TDM
  const bool issuer = (wave == 0);
  if (issuer) {
    tdm_load_tile_128x32(gA, lds_off32(&As[0][0]), K);
    tdm_load_tile_128x32(gB, lds_off32(&Bs[0][0]), K);
  }
#else
  const int rowL   = t >> 2;        // 0..63
  const int colSeg = (t & 3) * 8;   // 0,8,16,24 halves
  for (int rep = 0; rep < 2; rep++) {
    int r = rep * 64 + rowL;
    *(v8h*)&As[0][r * LDST + colSeg] = *(const v8h*)&gA[(size_t)r * K + colSeg];
    *(v8h*)&Bs[0][r * LDST + colSeg] = *(const v8h*)&gB[(size_t)r * K + colSeg];
  }
#endif

  for (int it = 0; it < nIter; it++) {
    const int cur = it & 1;
#if USE_TDM
    if (issuer) {
      if (it + 1 < nIter) {
        // prefetch next tile pair into the other buffer, keep it in flight
        tdm_load_tile_128x32(gA + (size_t)(it + 1) * 32, lds_off32(&As[cur ^ 1][0]), K);
        tdm_load_tile_128x32(gB + (size_t)(it + 1) * 32, lds_off32(&Bs[cur ^ 1][0]), K);
        __builtin_amdgcn_s_wait_tensorcnt(2);   // current buffer's 2 DMAs have landed
      } else {
        __builtin_amdgcn_s_wait_tensorcnt(0);
      }
    }
#endif
    __syncthreads();   // current buffer ready for all waves

    const int kfo = (lane >> 4) * 8;  // per-lane K base (ISA 16-bit A layout)
    const int rl  = lane & 15;
    v16h afrag[2], bfrag[4];
    for (int mi = 0; mi < 2; mi++) {
      int m = wm * 32 + mi * 16 + rl;
      union { v16h v; v8h h[2]; } u;
      u.h[0] = *(const v8h*)&As[cur][m * LDST + kfo];
      u.h[1] = *(const v8h*)&As[cur][m * LDST + kfo + 16];
      afrag[mi] = u.v;
    }
    for (int ni = 0; ni < 4; ni++) {
      int n = wn * 64 + ni * 16 + rl;
      union { v16h v; v8h h[2]; } u;
      u.h[0] = *(const v8h*)&Bs[cur][n * LDST + kfo];
      u.h[1] = *(const v8h*)&Bs[cur][n * LDST + kfo + 16];
      bfrag[ni] = u.v;
    }
    for (int mi = 0; mi < 2; mi++)
      for (int ni = 0; ni < 4; ni++)
        acc[mi][ni] = __builtin_amdgcn_wmma_f32_16x16x32_f16(
            false, afrag[mi], false, bfrag[ni], (short)0, acc[mi][ni], false, false);

#if !USE_TDM
    __syncthreads();
    if (it + 1 < nIter) {
      for (int rep = 0; rep < 2; rep++) {
        int r = rep * 64 + rowL;
        *(v8h*)&As[0][r * LDST + colSeg] = *(const v8h*)&gA[(size_t)r * K + (it+1)*32 + colSeg];
        *(v8h*)&Bs[0][r * LDST + colSeg] = *(const v8h*)&gB[(size_t)r * K + (it+1)*32 + colSeg];
      }
    }
#else
    __syncthreads();   // all waves done reading buffer `cur`; TDM may overwrite it next iter
#endif
  }

  // epilogue: D element (m,n) lives in vgpr r = m&7, lane = (n&15) + 16*(m>>3)
  const int ml = (lane >> 4) * 8;
  const int nl = lane & 15;
  const int seg = blockN >> 9;  // MODE_QKV: 0=q 1=k 2=v (block-uniform, scalar branch)
  for (int mi = 0; mi < 2; mi++)
    for (int ni = 0; ni < 4; ni++)
      for (int r = 0; r < 8; r++) {
        int gm = blockM + wm * 32 + mi * 16 + ml + r;
        int gn = blockN + wn * 64 + ni * 16 + nl;
        float v = acc[mi][ni][r];
        if (MODE == MODE_QKV) {
          v += bias[gn];
          if (seg == 0)      o16a[(size_t)gm * CCH + gn]              = (_Float16)(v * QSCALE); // Q (pre-scaled)
          else if (seg == 1) o16b[(size_t)gm * CCH + (gn - CCH)]      = (_Float16)v;            // K
          else               o16c[(size_t)(gn - 2*CCH) * SEQ + gm]    = (_Float16)v;            // V^T [C,S]
        } else if (MODE == MODE_SCORES) {
          ofl[(size_t)gm * SEQ + gn] = v;
        } else if (MODE == MODE_PV) {
          o16a[(size_t)gm * CCH + gn] = (_Float16)v;
        } else { // MODE_PROJ: out[b,o,s] = x + proj + bias
          size_t idx = (size_t)gn * SEQ + gm;
          ofl[idx] = resid[idx] + v + bias[gn];
        }
      }
}

// ---------------- row softmax: scores[1024] fp32 -> P fp16 ----------------
__global__ void softmax_kernel(const float* __restrict__ S, _Float16* __restrict__ P) {
  const int row = blockIdx.x;
  const int t   = threadIdx.x;
  const float4* sp = (const float4*)(S + (size_t)row * SEQ);
  float4 v = sp[t];
  __shared__ float red[256];
  red[t] = fmaxf(fmaxf(v.x, v.y), fmaxf(v.z, v.w));
  __syncthreads();
  for (int o = 128; o > 0; o >>= 1) { if (t < o) red[t] = fmaxf(red[t], red[t+o]); __syncthreads(); }
  const float rmax = red[0];
  __syncthreads();
  float e0 = __expf(v.x - rmax), e1 = __expf(v.y - rmax);
  float e2 = __expf(v.z - rmax), e3 = __expf(v.w - rmax);
  red[t] = e0 + e1 + e2 + e3;
  __syncthreads();
  for (int o = 128; o > 0; o >>= 1) { if (t < o) red[t] += red[t+o]; __syncthreads(); }
  const float inv = 1.f / red[0];
  _Float16* pp = P + (size_t)row * SEQ + t * 4;
  pp[0] = (_Float16)(e0 * inv);
  pp[1] = (_Float16)(e1 * inv);
  pp[2] = (_Float16)(e2 * inv);
  pp[3] = (_Float16)(e3 * inv);
}

// ---------------- launch ----------------
extern "C" void kernel_launch(void* const* d_in, const int* in_sizes, int n_in,
                              void* d_out, int out_size, void* d_ws, size_t ws_size,
                              hipStream_t stream) {
  const float* x      = (const float*)d_in[0];
  const float* gn_w   = (const float*)d_in[1];
  const float* gn_b   = (const float*)d_in[2];
  const float* qkv_w  = (const float*)d_in[3];
  const float* qkv_b  = (const float*)d_in[4];
  const float* proj_w = (const float*)d_in[5];
  const float* proj_b = (const float*)d_in[6];
  float* out = (float*)d_out;

  char* ws = (char*)d_ws;
  size_t off = 0;
  auto alloc = [&](size_t bytes) -> void* {
    void* p = ws + off;
    off += (bytes + 255) & ~(size_t)255;
    return p;
  };
  float*    stats = (float*)   alloc(BATCH * 2 * sizeof(float));
  _Float16* wq    = (_Float16*)alloc((size_t)THREEC * CCH * 2);
  _Float16* wp    = (_Float16*)alloc((size_t)CCH * CCH * 2);
  _Float16* ht    = (_Float16*)alloc((size_t)SEQ * CCH * 2);
  _Float16* qb    = (_Float16*)alloc((size_t)SEQ * CCH * 2);
  _Float16* kbf   = (_Float16*)alloc((size_t)SEQ * CCH * 2);
  _Float16* vt    = (_Float16*)alloc((size_t)CCH * SEQ * 2);
  _Float16* ao    = (_Float16*)alloc((size_t)SEQ * CCH * 2);
  _Float16* P     = (_Float16*)alloc((size_t)SEQ * SEQ * 2);
  float*    sc    = (float*)   alloc((size_t)SEQ * SEQ * 4);
  (void)ws_size; (void)in_sizes; (void)n_in; (void)out_size;

  gn_stats_kernel<<<BATCH, 256, 0, stream>>>(x, stats);
  convert_w_kernel<<<(THREEC*CCH + CCH*CCH) / 256, 256, 0, stream>>>(qkv_w, proj_w, wq, wp);

  for (int b = 0; b < BATCH; b++) {
    normalize_kernel<<<dim3(CCH/32, SEQ/32), dim3(32, 8), 0, stream>>>(x, gn_w, gn_b, stats, ht, b);
    // qkv: [S,3C] = ht[S,C] * wq[3C,C]^T (+bias); writes Q (scaled), K, V^T
    gemm_kernel<MODE_QKV><<<dim3(THREEC/128, SEQ/128), 256, 0, stream>>>(
        ht, wq, CCH, qkv_b, qb, kbf, vt, nullptr, nullptr);
    // scores: [S,S] = Q[S,C] * K[S,C]^T
    gemm_kernel<MODE_SCORES><<<dim3(SEQ/128, SEQ/128), 256, 0, stream>>>(
        qb, kbf, CCH, nullptr, nullptr, nullptr, nullptr, sc, nullptr);
    softmax_kernel<<<SEQ, 256, 0, stream>>>(sc, P);
    // attn out: [S,C] = P[S,S] * (V^T)[C,S]^T
    gemm_kernel<MODE_PV><<<dim3(CCH/128, SEQ/128), 256, 0, stream>>>(
        P, vt, SEQ, nullptr, ao, nullptr, nullptr, nullptr, nullptr);
    // proj + bias + residual, write [C,S] fp32 slab of output
    gemm_kernel<MODE_PROJ><<<dim3(CCH/128, SEQ/128), 256, 0, stream>>>(
        ao, wp, CCH, proj_b, nullptr, nullptr, nullptr,
        out + (size_t)b * CCH * SEQ, x + (size_t)b * CCH * SEQ);
  }
}
